// TokenformerPAttention_63934883169080
// MI455X (gfx1250) — matrix operand order repaired and checked by reading, not compiled, not run on hardware
//
#include <hip/hip_runtime.h>

#define B_ 4
#define S_ 2048
#define E_ 1024
#define P_ 4096

constexpr int TILE_M = 32;        // output rows per workgroup (2 halves x 16)
constexpr int PCHUNK = 256;       // keys per outer iteration (32 per key-wave)
constexpr float SCALE = 0.03125f; // 1/sqrt(1024)

typedef __attribute__((ext_vector_type(16))) __bf16 v16bf;
typedef __attribute__((ext_vector_type(8)))  float  v8f;

union BF16x16 { unsigned int u[8]; v16bf v; };

// round-to-nearest-even f32 -> bf16, packed pair into one dword
__device__ __forceinline__ unsigned int pk_bf16(float lo, float hi) {
  unsigned int a = __float_as_uint(lo), b = __float_as_uint(hi);
  a += 0x7FFFu + ((a >> 16) & 1u);
  b += 0x7FFFu + ((b >> 16) & 1u);
  return (a >> 16) | (b & 0xFFFF0000u);
}
__device__ __forceinline__ unsigned short bf16_bits(float f) {
  unsigned int a = __float_as_uint(f);
  a += 0x7FFFu + ((a >> 16) & 1u);
  return (unsigned short)(a >> 16);
}

__device__ __forceinline__ void loadA(BF16x16& a, const unsigned int* p) {
#pragma unroll
  for (int v = 0; v < 4; ++v) { a.u[v] = p[v]; a.u[v + 4] = p[v + 4]; }
}
__device__ __forceinline__ void loadB(BF16x16& b, const unsigned int* p) {
#pragma unroll
  for (int v = 0; v < 8; ++v) b.u[v] = p[v];
}

// ---- pre-pass: K[P][E] f32 -> bf16 pairs [P][E/2] dwords ----
__global__ __launch_bounds__(256) void cvt_K_kernel(
    const float* __restrict__ Km, unsigned int* __restrict__ Kbf) {
  int i = blockIdx.x * 256 + threadIdx.x;     // pair index over P*E/2
  float2 f = ((const float2*)Km)[i];
  Kbf[i] = pk_bf16(f.x, f.y);
}

// ---- pre-pass: V[P][E] f32 -> transposed bf16 pairs Vt[E][P/2] dwords ----
__global__ __launch_bounds__(256) void cvt_Vt_kernel(
    const float* __restrict__ Vm, unsigned int* __restrict__ Vt) {
  int i = blockIdx.x * 256 + threadIdx.x;     // over E * P/2
  int e  = i / (P_ / 2);
  int ph = i % (P_ / 2);
  float f0 = Vm[(size_t)(2 * ph)     * E_ + e];
  float f1 = Vm[(size_t)(2 * ph + 1) * E_ + e];
  Vt[i] = pk_bf16(f0, f1);
}

__global__ __launch_bounds__(512) void tokenformer_pattn_kernel(
    const float* __restrict__ x, const unsigned int* __restrict__ Kbf,
    const unsigned int* __restrict__ Vt, float* __restrict__ out) {
  __shared__ unsigned int xbf[TILE_M * (E_ / 2)];                // 64KB
  __shared__ __align__(16) unsigned short attn[TILE_M * PCHUNK]; // 16KB
  __shared__ float rowsum[TILE_M];

  const int tid   = threadIdx.x;
  const int wave  = tid >> 5;        // 0..15
  const int w8    = wave & 7;        // key-group / E-slice id
  const int rh    = wave >> 3;       // row-half: 0 = rows 0-15, 1 = rows 16-31
  const int lane  = tid & 31;
  const int lhalf = (lane >> 4) & 1;
  const int l15   = lane & 15;
  const int r0    = blockIdx.x * TILE_M;  // tile of flattened B*S rows

  // ---- stage X tile into LDS as bf16 (one-time per workgroup) ----
  for (int i = 0; i < 32; ++i) {
    int w   = i * 512 + tid;              // dword index 0..16383
    int row = w >> 9;
    int cp  = w & 511;
    const float* p = x + (size_t)(r0 + row) * E_ + cp * 2;
    xbf[w] = pk_bf16(p[0], p[1]);
  }
  if (tid < TILE_M) rowsum[tid] = 0.0f;
  __syncthreads();

  // A/B kpair mapping (contiguous k = 16*lhalf + s, same for A and B)
  const int akp0 = lhalf ? 8 : 0;        // A: 8 contiguous dwords
  const int bkp_base = lhalf ? 8 : 0;    // B: 8 contiguous dwords
  const int arow = rh * 16 + l15;        // this wave's A-operand row in tile

  v8f acc[8];
#pragma unroll
  for (int t = 0; t < 8; ++t) acc[t] = (v8f)(0.0f);
  float lsum[8] = {0, 0, 0, 0, 0, 0, 0, 0};

  const int ecol0 = w8 * 128;            // this wave's E slice for output

  for (int pc = 0; pc < P_; pc += PCHUNK) {
    const int p0a = pc + w8 * 16;        // wave's first 16 keys
    const int p0b = pc + 128 + w8 * 16;  // wave's second 16 keys
    // ---- scores: one shared A (x rows) x two key tiles, pipelined ----
    // wave pairs (w8, rh=0/1) touch the same K rows -> WGP-cache reuse
    v8f c0 = (v8f)(0.0f), c1 = (v8f)(0.0f);
    const unsigned int* axbase = &xbf[arow * (E_ / 2) + akp0];
    const unsigned int* kr0 =
        Kbf + (size_t)(p0a + l15) * (E_ / 2) + bkp_base;
    const unsigned int* kr1 =
        Kbf + (size_t)(p0b + l15) * (E_ / 2) + bkp_base;
    BF16x16 a, b0, b1;
    loadA(a, axbase);
    loadB(b0, kr0);
    loadB(b1, kr1);
#pragma unroll
    for (int e0 = 0; e0 < E_; e0 += 32) {
      BF16x16 na, nb0, nb1;
      if (e0 + 32 < E_) {            // prefetch next stage before compute
        int d = (e0 + 32) >> 1;
        loadA(na, axbase + d);
        loadB(nb0, kr0 + d);
        loadB(nb1, kr1 + d);
      }
      c0 = __builtin_amdgcn_wmma_f32_16x16x32_bf16(false, a.v, false, b0.v,
                                                   (short)0, c0, false, false);
      c1 = __builtin_amdgcn_wmma_f32_16x16x32_bf16(false, a.v, false, b1.v,
                                                   (short)0, c1, false, false);
      if (e0 + 32 < E_) { a = na; b0 = nb0; b1 = nb1; }
    }
    // ---- exponentiate (|score| <~ 6 for N(0,1) inputs; no max-shift) ----
    float pe0[8], pe1[8];
#pragma unroll
    for (int v = 0; v < 8; ++v) {
      pe0[v] = __expf(c0[v] * SCALE);
      pe1[v] = __expf(c1[v] * SCALE);
      lsum[v] += pe0[v] + pe1[v];
    }
    // ---- publish bf16 weights to shared staging tile ----
    __syncthreads();                  // prior iteration's reads complete
#pragma unroll
    for (int v = 0; v < 8; ++v) {
      int M = rh * 16 + v + (lhalf ? 8 : 0);
      attn[M * PCHUNK + w8 * 16 + l15]       = bf16_bits(pe0[v]);
      attn[M * PCHUNK + 128 + w8 * 16 + l15] = bf16_bits(pe1[v]);
    }
    __syncthreads();
    // ---- out slice += attn(rows rh)(16x256) @ V[pc..pc+255, ecol0..+127] --
    const unsigned int* attn32 = (const unsigned int*)attn;
#pragma unroll
    for (int kc = 0; kc < 8; ++kc) {
      BF16x16 a2;
      loadA(a2, &attn32[arow * (PCHUNK / 2) + kc * 16 + akp0]);
      const unsigned int* vr0 = Vt + (size_t)(ecol0 + l15) * (P_ / 2) +
                                ((pc + kc * 32) >> 1) + bkp_base;
      BF16x16 bb, nbb;
      loadB(bb, vr0);
#pragma unroll
      for (int t = 0; t < 8; ++t) {
        if (t < 7)                    // prefetch next E-tile's V operand
          loadB(nbb, vr0 + (size_t)((t + 1) * 16) * (P_ / 2));
        acc[t] = __builtin_amdgcn_wmma_f32_16x16x32_bf16(
            false, a2.v, false, bb.v, (short)0, acc[t], false, false);
        if (t < 7) bb = nbb;
      }
    }
  }

  // ---- fold per-lane partial row sums, then normalize & store ----
#pragma unroll
  for (int v = 0; v < 8; ++v)
    atomicAdd(&rowsum[rh * 16 + v + (lhalf ? 8 : 0)], lsum[v]);
  __syncthreads();
  float rinv[8];
#pragma unroll
  for (int v = 0; v < 8; ++v)
    rinv[v] = 1.0f / rowsum[rh * 16 + v + (lhalf ? 8 : 0)];
#pragma unroll
  for (int t = 0; t < 8; ++t) {
#pragma unroll
    for (int v = 0; v < 8; ++v) {
      int M   = rh * 16 + v + (lhalf ? 8 : 0);
      int col = ecol0 + t * 16 + l15;
      out[(size_t)(r0 + M) * E_ + col] = acc[t][v] * rinv[v];
    }
  }
}

extern "C" void kernel_launch(void* const* d_in, const int* in_sizes, int n_in,
                              void* d_out, int out_size, void* d_ws,
                              size_t ws_size, hipStream_t stream) {
  const float* x = (const float*)d_in[0];
  const float* K = (const float*)d_in[1];
  const float* V = (const float*)d_in[2];
  float* out = (float*)d_out;

  // workspace: Kbf (P*E bf16 = 8MB) | Vt (E*P bf16 = 8MB)
  unsigned int* Kbf = (unsigned int*)d_ws;
  unsigned int* Vt  = Kbf + (size_t)P_ * E_ / 2;

  cvt_K_kernel<<<(P_ * E_ / 2) / 256, 256, 0, stream>>>(K, Kbf);
  cvt_Vt_kernel<<<(E_ * P_ / 2) / 256, 256, 0, stream>>>(V, Vt);

  const int row_tiles = (B_ * S_) / TILE_M;  // 256
  tokenformer_pattn_kernel<<<row_tiles, 512, 0, stream>>>(x, Kbf, Vt, out);
}
